// Model_21285857919454
// MI455X (gfx1250) — matrix operand order repaired and checked by reading, compiled, not run on hardware
//
#include <hip/hip_runtime.h>
#include <hip/hip_bf16.h>

// ---------------------------------------------------------------------------
// CDNA5 (gfx1250): bf16 WMMA GEMMs (double-buffered LDS, vectorized staging)
//                  + atomic segment-sum + BN for the GNN pipeline
// ---------------------------------------------------------------------------

typedef __bf16 bf16_t;
typedef bf16_t v16bf __attribute__((ext_vector_type(16)));
typedef float  v8f   __attribute__((ext_vector_type(8)));
typedef unsigned int u32x4 __attribute__((ext_vector_type(4)));
typedef unsigned int u32x2 __attribute__((ext_vector_type(2)));

__device__ __forceinline__ unsigned short f2bf(float f) {
  // round-to-nearest-even f32 -> bf16
  unsigned int u = __float_as_uint(f);
  u += 0x7FFFu + ((u >> 16) & 1u);
  return (unsigned short)(u >> 16);
}

union FragBF {
  u32x4 u[2];
  v16bf v;
};

#define BM 256
#define BN 64
#define BK 32
#define LDA 40   // padded LDS row stride (ushorts) for A tile
#define LDB 72   // padded LDS row stride (ushorts) for B tile

// Stage one (BM x BK) A tile and (BK x BN) B tile into LDS as bf16.
__device__ __forceinline__ void stage_tile(
    const float* __restrict__ A1, int K1,
    const float* __restrict__ A2, int K2, int K,
    const float* __restrict__ B, int M, int N,
    int bM, int bN, int kb,
    unsigned short* __restrict__ As, unsigned short* __restrict__ Bs,
    int tid, bool vecA)
{
  if (vecA) {
    // K1, K2 multiples of 4: float4 chunks never straddle concat boundary.
    for (int i = tid; i < (BM * BK) / 4; i += 256) {
      int row = i >> 3;            // 8 float4 per 32-wide row
      int c4  = (i & 7) << 2;
      int gm = bM + row, gk0 = kb + c4;
      float x0 = 0.f, x1 = 0.f, x2 = 0.f, x3 = 0.f;
      if (gm < M && gk0 < K) {
        if (gk0 + 3 < K) {
          const float* p = (gk0 < K1) ? &A1[(size_t)gm * K1 + gk0]
                                      : &A2[(size_t)gm * K2 + (gk0 - K1)];
          float4 v = *(const float4*)p;
          x0 = v.x; x1 = v.y; x2 = v.z; x3 = v.w;
        } else {                    // K tail (unused for current shapes; safe)
          for (int e = 0; e < 4; ++e) {
            int gk = gk0 + e;
            float t = 0.f;
            if (gk < K)
              t = (gk < K1) ? A1[(size_t)gm * K1 + gk]
                            : A2[(size_t)gm * K2 + (gk - K1)];
            if (e == 0) x0 = t; else if (e == 1) x1 = t;
            else if (e == 2) x2 = t; else x3 = t;
          }
        }
      }
      u32x2 pk;
      pk[0] = (unsigned)f2bf(x0) | ((unsigned)f2bf(x1) << 16);
      pk[1] = (unsigned)f2bf(x2) | ((unsigned)f2bf(x3) << 16);
      *(u32x2*)&As[row * LDA + c4] = pk;
    }
  } else {
    // scalar path (embedding layer, K1 = 11)
    for (int i = tid; i < BM * BK; i += 256) {
      int row = i >> 5, kk = i & 31;
      int gm = bM + row, gk = kb + kk;
      float v = 0.f;
      if (gm < M && gk < K)
        v = (gk < K1) ? A1[(size_t)gm * K1 + gk]
                      : A2[(size_t)gm * K2 + (gk - K1)];
      As[row * LDA + kk] = f2bf(v);
    }
  }

  // B tile: N is a multiple of 4 for all layers.
  for (int i = tid; i < (BK * BN) / 4; i += 256) {
    int kk = i >> 4;               // 16 float4 per 64-wide row
    int c4 = (i & 15) << 2;
    int gk = kb + kk, gn0 = bN + c4;
    float x0 = 0.f, x1 = 0.f, x2 = 0.f, x3 = 0.f;
    if (gk < K && gn0 < N) {
      if (gn0 + 3 < N) {
        float4 v = *(const float4*)&B[(size_t)gk * N + gn0];
        x0 = v.x; x1 = v.y; x2 = v.z; x3 = v.w;
      } else {
        for (int e = 0; e < 4; ++e) {
          int gn = gn0 + e;
          float t = (gn < N) ? B[(size_t)gk * N + gn] : 0.f;
          if (e == 0) x0 = t; else if (e == 1) x1 = t;
          else if (e == 2) x2 = t; else x3 = t;
        }
      }
    }
    u32x2 pk;
    pk[0] = (unsigned)f2bf(x0) | ((unsigned)f2bf(x1) << 16);
    pk[1] = (unsigned)f2bf(x2) | ((unsigned)f2bf(x3) << 16);
    *(u32x2*)&Bs[kk * LDB + c4] = pk;
  }
}

// C[M,N] = concat(A1[M,K1], A2[M,K2]) @ B[K,N] + bias, optional ReLU.
// Block tile 256x64, 8 waves, wave tile 64x32 (4x2 WMMA), double-buffered LDS.
__global__ __launch_bounds__(256)
void gemm_wmma_bf16(const float* __restrict__ A1, int K1,
                    const float* __restrict__ A2, int K2,
                    const float* __restrict__ B,
                    const float* __restrict__ bias,
                    float* __restrict__ C,
                    int M, int N, int relu)
{
  __shared__ __align__(16) unsigned short As[2][BM * LDA];
  __shared__ __align__(16) unsigned short Bs[2][BK * LDB];

  const int tid  = threadIdx.x;
  const int lane = tid & 31;
  const int wid  = tid >> 5;
  const int wm   = wid & 3;     // 4 waves along M (64 rows each)
  const int wn   = wid >> 2;    // 2 waves along N (32 cols each)
  const int bM   = blockIdx.y * BM;
  const int bN   = blockIdx.x * BN;
  const int K    = K1 + K2;
  const bool vecA = ((K1 & 3) == 0) && ((K2 & 3) == 0);

  v8f acc[4][2];
  for (int i = 0; i < 4; ++i)
    for (int j = 0; j < 2; ++j)
      for (int r = 0; r < 8; ++r) acc[i][j][r] = 0.f;

  const int lhalf = lane >> 4;     // which 16-lane half
  const int mrow  = lane & 15;
  const int koff  = lhalf * 8;     // A-frag K interleave per ISA layout

  // prologue: stage first tile into buffer 0
  stage_tile(A1, K1, A2, K2, K, B, M, N, bM, bN, 0, As[0], Bs[0], tid, vecA);

  int buf = 0;
  for (int kb = 0; kb < K; kb += BK) {
    __syncthreads();   // staged tile [buf] visible; prior reads of [buf^1] done
    if (kb + BK < K)   // prefetch next tile into the other buffer
      stage_tile(A1, K1, A2, K2, K, B, M, N, bM, bN, kb + BK,
                 As[buf ^ 1], Bs[buf ^ 1], tid, vecA);

    // ---- fragments per ISA VGPR layout, 8 WMMAs ----
    FragBF fa[4], fb[2];
    const unsigned short* as = As[buf];
    const unsigned short* bs = Bs[buf];
    for (int mi = 0; mi < 4; ++mi) {
      int m = wm * 64 + mi * 16 + mrow;
      // elements 0..7  = A[m][koff .. koff+7]
      // elements 8..15 = A[m][koff+16 .. koff+23]
      fa[mi].u[0] = *(const u32x4*)&as[m * LDA + koff];
      fa[mi].u[1] = *(const u32x4*)&as[m * LDA + koff + 16];
    }
    for (int ni = 0; ni < 2; ++ni) {
      int nb = wn * 32 + ni * 16;  // B: lane = N column, K per A-frag interleave
      const unsigned short* p = &bs[koff * LDB + nb + mrow];
      for (int q = 0; q < 4; ++q) {
        fb[ni].u[0][q] = (unsigned)p[(2 * q) * LDB] | ((unsigned)p[(2 * q + 1) * LDB] << 16);
        fb[ni].u[1][q] = (unsigned)p[(16 + 2 * q) * LDB] | ((unsigned)p[(16 + 2 * q + 1) * LDB] << 16);
      }
    }
    for (int mi = 0; mi < 4; ++mi)
      for (int ni = 0; ni < 2; ++ni)
        acc[mi][ni] = __builtin_amdgcn_wmma_f32_16x16x32_bf16(
            false, fa[mi].v, false, fb[ni].v,
            (short)0, acc[mi][ni], false, false);
    buf ^= 1;
  }

  // ---- epilogue: C/D layout = lane(N), vgpr r (M=r or r+8), + bias/relu ----
  for (int mi = 0; mi < 4; ++mi)
    for (int ni = 0; ni < 2; ++ni)
      for (int r = 0; r < 8; ++r) {
        int grow = bM + wm * 64 + mi * 16 + lhalf * 8 + r;
        int gcol = bN + wn * 32 + ni * 16 + (lane & 15);
        if (grow < M && gcol < N) {
          float v = acc[mi][ni][r] + bias[gcol];
          if (relu) v = fmaxf(v, 0.f);
          C[(size_t)grow * N + gcol] = v;
        }
      }
}

// ---------------------------------------------------------------------------
__global__ void zero_kernel(float* __restrict__ p, size_t n) {
  size_t i = (size_t)blockIdx.x * blockDim.x + threadIdx.x;
  if (i < n) p[i] = 0.f;
}

// segment_sum: out[dst[e], :] += h[src[e], :]
__global__ void agg_kernel(const float* __restrict__ h,
                           const int* __restrict__ src,
                           const int* __restrict__ dst,
                           float* __restrict__ out, int E, int C4) {
  long long t = (long long)blockIdx.x * blockDim.x + threadIdx.x;
  if (t >= (long long)E * C4) return;
  int e = (int)(t / C4);
  int c = (int)(t % C4) * 4;
  int Cf = C4 * 4;
  int s = src[e], d = dst[e];
  const float4 v = *(const float4*)&h[(size_t)s * Cf + c];
  float* o = &out[(size_t)d * Cf + c];
  atomicAdd(o + 0, v.x);
  atomicAdd(o + 1, v.y);
  atomicAdd(o + 2, v.z);
  atomicAdd(o + 3, v.w);
}

// one block per channel: mean + rsqrt(var + eps)
__global__ void bn_stats(const float* __restrict__ h, int M, int C,
                         float* __restrict__ mean, float* __restrict__ rstd) {
  int ch = blockIdx.x;
  int tid = threadIdx.x;
  float s = 0.f, s2 = 0.f;
  for (int r = tid; r < M; r += blockDim.x) {
    float v = h[(size_t)r * C + ch];
    s += v; s2 += v * v;
  }
  __shared__ float sh[256], sh2[256];
  sh[tid] = s; sh2[tid] = s2;
  __syncthreads();
  for (int o = 128; o > 0; o >>= 1) {
    if (tid < o) { sh[tid] += sh[tid + o]; sh2[tid] += sh2[tid + o]; }
    __syncthreads();
  }
  if (tid == 0) {
    float mu = sh[0] / (float)M;
    float var = sh2[0] / (float)M - mu * mu;
    mean[ch] = mu;
    rstd[ch] = rsqrtf(var + 1e-5f);
  }
}

__global__ void bn_apply(float* __restrict__ h,
                         const float* __restrict__ g, const float* __restrict__ b,
                         const float* __restrict__ mean, const float* __restrict__ rstd,
                         size_t n, int C) {
  size_t i = (size_t)blockIdx.x * blockDim.x + threadIdx.x;
  if (i < n) {
    int c = (int)(i % C);
    h[i] = g[c] * (h[i] - mean[c]) * rstd[c] + b[c];
  }
}

// ---------------------------------------------------------------------------
extern "C" void kernel_launch(void* const* d_in, const int* in_sizes, int n_in,
                              void* d_out, int out_size, void* d_ws, size_t ws_size,
                              hipStream_t stream) {
  const int N = 50000, E = 800000;

  const float* x    = (const float*)d_in[0];
  const int*   esrc = (const int*)d_in[1];
  const int*   edst = (const int*)d_in[2];
  // d_in[3] = batch (unused, single graph)
  const float* embW = (const float*)d_in[4];
  const float* embb = (const float*)d_in[5];
  struct ConvP { const float *W1, *b1, *g1, *bt1, *W2, *b2; };
  ConvP conv[4];
  for (int i = 0; i < 4; ++i) {
    int base = 6 + i * 6;
    conv[i] = { (const float*)d_in[base + 0], (const float*)d_in[base + 1],
                (const float*)d_in[base + 2], (const float*)d_in[base + 3],
                (const float*)d_in[base + 4], (const float*)d_in[base + 5] };
  }
  const float* g   = (const float*)d_in[30];
  const float* bt  = (const float*)d_in[31];
  const float* l1W = (const float*)d_in[32]; const float* l1b = (const float*)d_in[33];
  const float* l2W = (const float*)d_in[34]; const float* l2b = (const float*)d_in[35];
  const float* l3W = (const float*)d_in[36]; const float* l3b = (const float*)d_in[37];

  float* ws   = (float*)d_ws;
  float* buf0 = ws;                               // N*600
  float* buf1 = buf0 + (size_t)N * 600;           // N*600
  float* bufA = buf1 + (size_t)N * 600;           // N*256 (agg ping)
  float* bufB = bufA + (size_t)N * 256;           // N*256 (agg pong)
  float* bufZ = bufB + (size_t)N * 256;           // N*128 (MLP hidden)
  float* mu   = bufZ + (size_t)N * 128;           // 600
  float* rs   = mu + 600;                         // 600

  auto gemm = [&](const float* A1, int K1, const float* A2, int K2,
                  const float* B, const float* bias, float* C, int M, int Nc, int relu) {
    dim3 grid((Nc + BN - 1) / BN, (M + BM - 1) / BM);
    gemm_wmma_bf16<<<grid, dim3(256), 0, stream>>>(A1, K1, A2, K2, B, bias, C, M, Nc, relu);
  };

  auto segsum = [&](const float* hin, float* out, int C) {
    size_t n = (size_t)N * C;
    zero_kernel<<<(unsigned)((n + 255) / 256), 256, 0, stream>>>(out, n);
    int C4 = C / 4;
    long long total = (long long)E * C4;
    agg_kernel<<<(unsigned)((total + 255) / 256), 256, 0, stream>>>(hin, esrc, edst, out, E, C4);
  };

  auto do_conv = [&](const float* in, int Cin, const ConvP& p, int Cout, int hops, float* out) {
    const float* a = in;
    float* ping = bufA; float* pong = bufB;
    for (int h = 0; h < hops; ++h) {
      segsum(a, ping, Cin);
      a = ping;
      float* t = ping; ping = pong; pong = t;
    }
    // z = relu(concat(h, a) @ W1 + b1)
    gemm(in, Cin, a, Cin, p.W1, p.b1, bufZ, N, 128, 1);
    // z = bn(z)
    bn_stats<<<128, 256, 0, stream>>>(bufZ, N, 128, mu, rs);
    bn_apply<<<(unsigned)(((size_t)N * 128 + 255) / 256), 256, 0, stream>>>(
        bufZ, p.g1, p.bt1, mu, rs, (size_t)N * 128, 128);
    // out = z @ W2 + b2
    gemm(bufZ, 128, nullptr, 0, p.W2, p.b2, out, N, Cout, 0);
  };

  // h = x @ embW + embb
  gemm(x, 11, nullptr, 0, embW, embb, buf0, N, 64, 0);

  do_conv(buf0,  64, conv[0], 256, 1, buf1);
  do_conv(buf1, 256, conv[1], 256, 2, buf0);
  do_conv(buf0, 256, conv[2], 256, 3, buf1);
  do_conv(buf1, 256, conv[3], 600, 4, buf0);

  // final BN over 600 channels
  bn_stats<<<600, 256, 0, stream>>>(buf0, N, 600, mu, rs);
  bn_apply<<<(unsigned)(((size_t)N * 600 + 255) / 256), 256, 0, stream>>>(
      buf0, g, bt, mu, rs, (size_t)N * 600, 600);

  // dense head
  gemm(buf0, 600, nullptr, 0, l1W, l1b, buf1, N, 600, 1);
  gemm(buf1, 600, nullptr, 0, l2W, l2b, buf0, N, 600, 1);
  gemm(buf0, 600, nullptr, 0, l3W, l3b, (float*)d_out, N, 600, 0);
}
